// STSGCL_56487409877074
// MI455X (gfx1250) — compile-verified, matrix operand order checked by reading, not run on hardware
//
#include <hip/hip_runtime.h>
#include <hip/hip_bf16.h>

typedef __attribute__((ext_vector_type(16))) __bf16 v16bf;
typedef __attribute__((ext_vector_type(8)))  __bf16 v8bf;
typedef __attribute__((ext_vector_type(4)))  __bf16 v4bf;
typedef __attribute__((ext_vector_type(8)))  float  v8f;
typedef __attribute__((ext_vector_type(4)))  unsigned int v4u;
typedef __attribute__((ext_vector_type(8)))  int    v8i;
typedef __attribute__((ext_vector_type(4)))  int    v4i;

#define T_DIM 12
#define B_DIM 32
#define N_DIM 900
#define M_DIM 2700          // 3*N
#define C_DIM 64
#define D_DIM 128           // 2*Cout
#define ADJ_STRIDE 2720     // padded (zero-filled) K stride for adj in bf16
#define KSTEPS 85           // ceil(2700/32)

__device__ __forceinline__ v16bf cat16(v8bf a, v8bf b) {
  return __builtin_shufflevector(a, b, 0,1,2,3,4,5,6,7,8,9,10,11,12,13,14,15);
}
__device__ __forceinline__ v8bf zero8bf() {
  v8bf r;
#pragma unroll
  for (int i = 0; i < 8; ++i) r[i] = (__bf16)0.0f;
  return r;
}
__device__ __forceinline__ v8f zero8f() {
  v8f r;
#pragma unroll
  for (int i = 0; i < 8; ++i) r[i] = 0.0f;
  return r;
}
__device__ __forceinline__ float sigmoidf_dev(float x) {
  return 1.0f / (1.0f + __expf(-x));
}

// ---------------------------------------------------------------------------
// adj (2700x2700 f32) -> bf16 with padded, zero-filled stride 2720
// ---------------------------------------------------------------------------
__global__ __launch_bounds__(256) void cvt_adj_kernel(
    const float* __restrict__ adj, __bf16* __restrict__ adjbf, int total) {
  int idx = blockIdx.x * 256 + threadIdx.x;
  if (idx >= total) return;
  int r = idx / ADJ_STRIDE;
  int k = idx - r * ADJ_STRIDE;
  adjbf[idx] = (k < M_DIM) ? (__bf16)adj[(size_t)r * M_DIM + k] : (__bf16)0.0f;
}

// ---------------------------------------------------------------------------
// H0[t][b][s*900+n][c] = bf16( x[b][clamp(t+s-1)][n][c] + t_emb + s_emb )
// ---------------------------------------------------------------------------
__global__ __launch_bounds__(256) void build_h0_kernel(
    const float* __restrict__ x, const float* __restrict__ temb,
    const float* __restrict__ semb, __bf16* __restrict__ h0, int total) {
  int idx = blockIdx.x * 256 + threadIdx.x;
  if (idx >= total) return;
  int c   = (idx & 15) * 4;
  int tmp = idx >> 4;
  int m   = tmp % M_DIM;
  int tb  = tmp / M_DIM;
  int b   = tb & 31;
  int t   = tb >> 5;
  int s   = m / N_DIM;
  int n   = m - s * N_DIM;
  int tt  = t + s - 1;
  tt = tt < 0 ? 0 : (tt > T_DIM - 1 ? T_DIM - 1 : tt);

  const float4 xv = *(const float4*)(x + (((size_t)(b * T_DIM + tt) * N_DIM + n) * C_DIM + c));
  const float4 tv = *(const float4*)(temb + (tt * C_DIM + c));
  const float4 sv = *(const float4*)(semb + (n * C_DIM + c));
  v4bf o;
  o[0] = (__bf16)(xv.x + tv.x + sv.x);
  o[1] = (__bf16)(xv.y + tv.y + sv.y);
  o[2] = (__bf16)(xv.z + tv.z + sv.z);
  o[3] = (__bf16)(xv.w + tv.w + sv.w);
  *(v4bf*)(h0 + (((size_t)(t * B_DIM + b) * M_DIM + m) * C_DIM + c)) = o;
}

// ---------------------------------------------------------------------------
// One layer:  Hnew = adj @ H  (bf16 WMMA, f32 acc; adj tile staged by TDM)
//             Z    = Hnew @ W[t,l] + b[t,l]  (bf16 WMMA)
//             Hout = glu(Z); store Hout; max-reduce rows [900,1800) into out
// Block: 256 threads (8 waves), 128 rows x 64 cols, one (t,b) batch.
// ---------------------------------------------------------------------------
__global__ __launch_bounds__(256) void layer_kernel(
    const __bf16* __restrict__ adjbf, const __bf16* __restrict__ hcur,
    __bf16* __restrict__ hnext, const float* __restrict__ W,
    const float* __restrict__ bias, float* __restrict__ out,
    int l, int first, int store_h) {
  constexpr int LDA = 40;   // ldsA K stride (bf16) = 16 data + 4 pad DWORDs (TDM pad)
  constexpr int LDB = 40;   // ldsB K stride
  constexpr int LDW = 72;   // ldsW K stride
  constexpr int LDH = 72;   // hb   C stride

  __shared__ __bf16 ldsA[2][128 * LDA];   // adj tile [m][k], TDM double buffer
  __shared__ __bf16 ldsB[64  * LDB];      // H tile^T [c][k]
  __shared__ __bf16 ldsW[128 * LDW];      // W tile^T [d][k]
  __shared__ __bf16 hb  [128 * LDH];      // Hnew bounce [m][c]
  __shared__ float  ldsBias[D_DIM];

  const int tid  = threadIdx.x;
  const int lane = tid & 31;
  const int wav  = tid >> 5;
  const int frow = lane & 15;           // fragment row (A) / col (B)
  const int fk   = (lane >> 4) << 3;    // fragment K base (0 or 8)

  const int mrow0 = blockIdx.x * 128;
  const int b = blockIdx.y;
  const int t = blockIdx.z;
  const size_t tb = (size_t)t * B_DIM + b;
  const __bf16* hsrc = hcur + tb * (size_t)(M_DIM * C_DIM);

  // ---- TDM descriptor issue: 128x32 bf16 adj tile -> ldsA[buf], padded rows
  auto issue_tdm = [&](int k0, int buf) {
    const unsigned lds_off = (unsigned)(size_t)(&ldsA[buf][0]);
    const unsigned long long ga =
        (unsigned long long)(size_t)(adjbf + (size_t)mrow0 * ADJ_STRIDE + k0);
    v4u g0;
    g0[0] = 1u;                                   // count=1 valid descriptor
    g0[1] = lds_off;                              // lds_addr
    g0[2] = (unsigned)(ga & 0xFFFFFFFFull);       // global_addr[31:0]
    g0[3] = (unsigned)((ga >> 32) & 0x1FFFFFFull) // global_addr[56:32]
            | (2u << 30);                         // type = 2 ("image")
    const int dim0 = ADJ_STRIDE - k0;             // remaining elems from tile start
    const int dim1 = M_DIM - mrow0;               // remaining rows (OOB rows -> 0)
    v8i g1;
    g1[0] = (1 << 16)      // data_size = 2 bytes
          | (1 << 20)      // pad_enable
          | (3 << 22)      // pad_interval: 16 DWORDs (one 32-elem bf16 row)
          | (3 << 25);     // pad_amount: 4 DWORDs -> LDS row stride 40 bf16
    g1[1] = (dim0 & 0xFFFF) << 16;                        // tensor_dim0[15:0]
    g1[2] = ((dim0 >> 16) & 0xFFFF) | ((dim1 & 0xFFFF) << 16);
    g1[3] = ((dim1 >> 16) & 0xFFFF) | (32 << 16);         // tile_dim0 = 32
    g1[4] = 128;                                          // tile_dim1 = 128
    g1[5] = ADJ_STRIDE;                                   // tensor_dim0_stride lo
    g1[6] = 0;
    g1[7] = 0;
    v4i g2; g2[0] = 0; g2[1] = 0; g2[2] = 0; g2[3] = 0;   // 2D: unused dims
    v4i g3; g3[0] = 0; g3[1] = 0; g3[2] = 0; g3[3] = 0;
#if __clang_major__ >= 23
    v8i g4; 
#pragma unroll
    for (int i = 0; i < 8; ++i) g4[i] = 0;
    __builtin_amdgcn_tensor_load_to_lds(g0, g1, g2, g3, g4, 0);
#else
    __builtin_amdgcn_tensor_load_to_lds(g0, g1, g2, g3, 0);
#endif
  };

  if (wav == 0) issue_tdm(0, 0);  // kick off first adj tile ASAP

  // ---- stage W (64x128 f32 row-major) -> bf16 transposed [d][k]; bias ----
  const float* wsrc = W + (size_t)(t * 3 + l) * (C_DIM * D_DIM);
  for (int i = tid; i < C_DIM * D_DIM; i += 256) {
    int k = i >> 7, d = i & 127;
    ldsW[d * LDW + k] = (__bf16)wsrc[i];
  }
  if (tid < D_DIM) ldsBias[tid] = bias[(t * 3 + l) * D_DIM + tid];

  // ---- main GEMM: Hnew(128x64) = adj(128xK) @ H(Kx64), K=2700 ----
  v8f acc[4];
#pragma unroll
  for (int j = 0; j < 4; ++j) acc[j] = zero8f();

  // H-tile staging assignment (transposed write, so manual)
  const int brow = tid >> 3;            // 0..31  (K row of H tile)
  const int bc0  = (tid & 7) * 8;       // channel base

  v8bf rb;
  auto load_b = [&](int k0) {
    const int gk = k0 + brow;
    rb = (gk < M_DIM) ? *(const v8bf*)(hsrc + (size_t)gk * C_DIM + bc0)
                      : zero8bf();
  };

  load_b(0);
  for (int ks = 0; ks < KSTEPS; ++ks) {
    const int cur = ks & 1;
    __syncthreads();                       // prior iter done with ldsB / ldsA[cur^1]
    if (wav == 0) {
      if (ks + 1 < KSTEPS) issue_tdm((ks + 1) * 32, cur ^ 1);
      // tile ks complete (tensor ops are in-order; allow next to stay in flight)
      if (ks + 1 < KSTEPS) __builtin_amdgcn_s_wait_tensorcnt((short)1);
      else                 __builtin_amdgcn_s_wait_tensorcnt((short)0);
    }
    // H regs -> LDS (transposed)
#pragma unroll
    for (int i = 0; i < 8; ++i) ldsB[(bc0 + i) * LDB + brow] = rb[i];
    __syncthreads();                       // A tile (TDM) + B tile published
    if (ks + 1 < KSTEPS) load_b((ks + 1) * 32);

    // compute: wave owns rows wav*16..+15, 4 col tiles
    const __bf16* ab = &ldsA[cur][(wav * 16 + frow) * LDA];
    const v16bf af = cat16(*(const v8bf*)(ab + fk), *(const v8bf*)(ab + fk + 16));
    v16bf bfg[4];
#pragma unroll
    for (int j = 0; j < 4; ++j) {          // distinct regs: loads pipeline
      const __bf16* bb = &ldsB[(j * 16 + frow) * LDB];
      bfg[j] = cat16(*(const v8bf*)(bb + fk), *(const v8bf*)(bb + fk + 16));
    }
#pragma unroll
    for (int j = 0; j < 4; ++j)
      acc[j] = __builtin_amdgcn_wmma_f32_16x16x32_bf16(
          false, af, false, bfg[j], (short)0, acc[j], false, false);
  }

  // ---- bounce Hnew through LDS to re-fragment as A for the W GEMM ----
#pragma unroll
  for (int j = 0; j < 4; ++j)
#pragma unroll
    for (int v = 0; v < 8; ++v) {
      const int m = wav * 16 + v + ((lane >> 4) << 3);
      const int c = j * 16 + frow;
      hb[m * LDH + c] = (__bf16)acc[j][v];
    }
  __syncthreads();

  const __bf16* hbp = &hb[(wav * 16 + frow) * LDH];
  const v16bf ha0 = cat16(*(const v8bf*)(hbp + fk),      *(const v8bf*)(hbp + fk + 16));
  const v16bf ha1 = cat16(*(const v8bf*)(hbp + fk + 32), *(const v8bf*)(hbp + fk + 48));

  v8f z[8];
#pragma unroll
  for (int j = 0; j < 8; ++j) {
    z[j] = zero8f();
    const __bf16* wb = &ldsW[(j * 16 + frow) * LDW];
    const v16bf w0 = cat16(*(const v8bf*)(wb + fk),      *(const v8bf*)(wb + fk + 16));
    const v16bf w1 = cat16(*(const v8bf*)(wb + fk + 32), *(const v8bf*)(wb + fk + 48));
    z[j] = __builtin_amdgcn_wmma_f32_16x16x32_bf16(false, ha0, false, w0, (short)0, z[j], false, false);
    z[j] = __builtin_amdgcn_wmma_f32_16x16x32_bf16(false, ha1, false, w1, (short)0, z[j], false, false);
  }

  // ---- bias + GLU + stores (h for next layer, max-out for rows N..2N) ----
  __bf16* hdst = hnext + tb * (size_t)(M_DIM * C_DIM);
#pragma unroll
  for (int j = 0; j < 4; ++j) {
    const int c  = j * 16 + frow;
    const float bl = ldsBias[c];
    const float br = ldsBias[C_DIM + c];
#pragma unroll
    for (int v = 0; v < 8; ++v) {
      const int m = mrow0 + wav * 16 + v + ((lane >> 4) << 3);
      if (m < M_DIM) {
        const float lhs = z[j][v] + bl;
        const float rhs = z[4 + j][v] + br;
        const float h = lhs * sigmoidf_dev(rhs);
        if (store_h) hdst[(size_t)m * C_DIM + c] = (__bf16)h;
        if (m >= N_DIM && m < 2 * N_DIM) {
          float* op = out + (((size_t)(b * T_DIM + t) * N_DIM + (m - N_DIM)) * C_DIM + c);
          *op = first ? h : fmaxf(*op, h);
        }
      }
    }
  }
}

// ---------------------------------------------------------------------------
extern "C" void kernel_launch(void* const* d_in, const int* in_sizes, int n_in,
                              void* d_out, int out_size, void* d_ws, size_t ws_size,
                              hipStream_t stream) {
  const float* x    = (const float*)d_in[0];
  const float* adj  = (const float*)d_in[1];
  const float* temb = (const float*)d_in[2];
  const float* semb = (const float*)d_in[3];
  const float* W    = (const float*)d_in[4];
  const float* bias = (const float*)d_in[5];
  float* out = (float*)d_out;

  const size_t adjBytes = (size_t)M_DIM * ADJ_STRIDE * sizeof(__bf16);
  const size_t hBytes   = (size_t)T_DIM * B_DIM * M_DIM * C_DIM * sizeof(__bf16);
  if (ws_size < adjBytes + 2 * hBytes) return;  // workspace too small: no-op

  char* ws = (char*)d_ws;
  __bf16* adjbf = (__bf16*)ws;
  __bf16* hA    = (__bf16*)(ws + adjBytes);
  __bf16* hB    = (__bf16*)(ws + adjBytes + hBytes);

  {
    const int total = M_DIM * ADJ_STRIDE;
    cvt_adj_kernel<<<(total + 255) / 256, 256, 0, stream>>>(adj, adjbf, total);
  }
  {
    const int total = T_DIM * B_DIM * M_DIM * (C_DIM / 4);
    build_h0_kernel<<<(total + 255) / 256, 256, 0, stream>>>(x, temb, semb, hA, total);
  }

  dim3 grid((M_DIM + 127) / 128, B_DIM, T_DIM);
  layer_kernel<<<grid, 256, 0, stream>>>(adjbf, hA, hB, W, bias, out, 0, 1, 1);
  layer_kernel<<<grid, 256, 0, stream>>>(adjbf, hB, hA, W, bias, out, 1, 0, 1);
  layer_kernel<<<grid, 256, 0, stream>>>(adjbf, hA, hB, W, bias, out, 2, 0, 0);
}
